// MultiHeadAttention_89197880803467
// MI455X (gfx1250) — matrix operand order repaired and checked by reading, not compiled, not run on hardware
//
#include <hip/hip_runtime.h>
#include <hip/hip_bf16.h>
#include <math.h>

// ---------------------------------------------------------------------------
// Triangle attention (AlphaFold-style) for MI455X / gfx1250, wave32 + WMMA.
//   C=32, N_HEAD=8, C_IN=128, B=1, S=256
// GEMMs via v_wmma_f32_16x16x32_bf16; LDS staging via TDM tensor_load_to_lds.
// ---------------------------------------------------------------------------

typedef __bf16 bf16_t;
typedef __attribute__((ext_vector_type(4)))  __bf16 v4bf;
typedef __attribute__((ext_vector_type(8)))  __bf16 v8bf;
typedef __attribute__((ext_vector_type(16))) __bf16 v16bf;
typedef __attribute__((ext_vector_type(8)))  float  v8f;
typedef __attribute__((ext_vector_type(4)))  unsigned int u32x4;
typedef __attribute__((ext_vector_type(4)))  int i32x4;
typedef __attribute__((ext_vector_type(8)))  int i32x8;

#define S_DIM   256
#define NHEAD   8
#define CH      32      // head dim
#define CIN     128
#define NSH     (S_DIM * NHEAD)          // 2048 (s,h) blocks
#define QK_SCALE 0.17677669529663687f    // 1/sqrt(32)

__device__ __forceinline__ v8f wmma_bf16(v16bf a, v16bf b, v8f c) {
  // D = A(16x32) * B(32x16) + C, fp32 accumulate
  return __builtin_amdgcn_wmma_f32_16x16x32_bf16(
      /*neg_a=*/false, a, /*neg_b=*/false, b,
      /*c_mod=*/(short)0, c, /*reuse_a=*/false, /*reuse_b=*/false);
}

// Byte address of an LDS object (AS3 offset) for TDM descriptors.
__device__ __forceinline__ unsigned lds_addr_of(const void* p) {
  return (unsigned)(unsigned long long)(__attribute__((address_space(3))) const void*)p;
}

// TDM: contiguous 1-D copy of nelem 2-byte elements, global -> LDS.
// D# built per CDNA5 ISA 8.3/8.4: group0 {count=1, lds_addr, global_addr,
// type=2}; group1 {data_size=2B, tensor_dim0=tile_dim0=nelem, dim1=1,
// dim0_stride=nelem}; groups 2/3 zero (<=2D tensor).
__device__ __forceinline__ void tdm_load_1d_bf16(const void* gsrc, unsigned lds_byte_addr,
                                                 unsigned nelem) {
  unsigned long long ga = (unsigned long long)(size_t)gsrc;
  u32x4 g0;
  g0[0] = 1u;                                           // count=1 (valid), user mode
  g0[1] = lds_byte_addr;                                // lds_addr
  g0[2] = (unsigned)(ga & 0xffffffffu);                 // global_addr[31:0]
  g0[3] = (unsigned)((ga >> 32) & 0x01ffffffu) | (2u << 30);  // addr[56:32] | type=2
  i32x8 g1;
  g1[0] = (int)(1u << 16);                              // data_size=1 (2 bytes), mask=0
  g1[1] = (int)((nelem & 0xffffu) << 16);               // tensor_dim0[15:0] @ bit48
  g1[2] = (int)((nelem >> 16) & 0xffffu) | (1 << 16);   // tensor_dim0 hi | tensor_dim1=1
  g1[3] = (int)((nelem & 0xffffu) << 16);               // tile_dim0 @ bit112 (nelem<=65535)
  g1[4] = 1;                                            // tile_dim1=1
  g1[5] = (int)nelem;                                   // tensor_dim0_stride[31:0]
  g1[6] = 0;
  g1[7] = 0;
  i32x4 z4 = {0, 0, 0, 0};
#if __clang_major__ >= 23
  i32x8 z8 = {0, 0, 0, 0, 0, 0, 0, 0};
  __builtin_amdgcn_tensor_load_to_lds(g0, g1, z4, z4, z8, 0);
#else
  __builtin_amdgcn_tensor_load_to_lds(g0, g1, z4, z4, 0);
#endif
}

// A fragment (16x32 bf16, M x K), row-major source with leading dim ld.
// Per CDNA5 ISA 7.12.2: lane m = lane&15; half = lane>>4;
//   elems 0..7  -> K = k0 + half*8 + e
//   elems 8..15 -> K = k0 + 16 + half*8 + (e-8)
__device__ __forceinline__ v16bf load_a_frag(const bf16_t* base, int ld, int m0, int k0) {
  const int lane = threadIdx.x & 31;
  const int hl   = lane >> 4;
  const bf16_t* p = base + (size_t)(m0 + (lane & 15)) * ld + k0 + hl * 8;
  v8bf lo = *(const v8bf*)(p);
  v8bf hi = *(const v8bf*)(p + 16);
  v16bf r;
#pragma unroll
  for (int i = 0; i < 8; ++i) { r[i] = lo[i]; r[i + 8] = hi[i]; }
  return r;
}

// B fragment (32x16 bf16, K x N) from K-major storage: base[n*ldk + k].
// Per CDNA5 ISA: N = n0 + lane&15; elems 0..7 -> K = k0 + 8*(lane>>4) + e, elems 8..15 -> K = k0 + 16 + 8*(lane>>4) + (e-8).
__device__ __forceinline__ v16bf load_b_frag_kmajor(const bf16_t* base, int ldk, int k0, int n0) {
  const int lane = threadIdx.x & 31;
  const int hl   = lane >> 4;
  const bf16_t* p = base + (size_t)(n0 + (lane & 15)) * ldk + k0 + hl * 8;
  v8bf lo = *(const v8bf*)(p);
  v8bf hi = *(const v8bf*)(p + 16);
  v16bf r;
#pragma unroll
  for (int i = 0; i < 8; ++i) { r[i] = lo[i]; r[i + 8] = hi[i]; }
  return r;
}

// ---------------------------------------------------------------------------
// Kernel 1a: cast x (65536x128 f32) -> bf16
// ---------------------------------------------------------------------------
__global__ void __launch_bounds__(256) cast_x_kernel(const float* __restrict__ x,
                                                     bf16_t* __restrict__ xb, int n4) {
  int i = blockIdx.x * 256 + threadIdx.x;
  if (i < n4) {
    float4 v = ((const float4*)x)[i];
    v4bf o = { (bf16_t)v.x, (bf16_t)v.y, (bf16_t)v.z, (bf16_t)v.w };
    ((v4bf*)xb)[i] = o;
  }
}

// ---------------------------------------------------------------------------
// Kernel 1b: repack weights K-major, bf16.
//   Wt  [1024][128] : columns of [Wq|Wk|Wv|Wg] (each CIN x 256, row-major)
//   Wot [128][256]  : columns of Wo (256 x 128, row-major)
// ---------------------------------------------------------------------------
__global__ void __launch_bounds__(256) prep_w_kernel(
    const float* __restrict__ Wq, const float* __restrict__ Wk,
    const float* __restrict__ Wv, const float* __restrict__ Wg,
    const float* __restrict__ Wo, bf16_t* __restrict__ Wt, bf16_t* __restrict__ Wot) {
  int idx = blockIdx.x * 256 + threadIdx.x;
  if (idx < 1024 * 128) {
    int j = idx >> 7, c = idx & 127;           // Wt[j][c] = W[c][j]
    const float* W = (j < 256) ? Wq : (j < 512) ? Wk : (j < 768) ? Wv : Wg;
    Wt[idx] = (bf16_t)W[c * 256 + (j & 255)];
  }
  if (idx < 128 * 256) {
    int n = idx >> 8, k = idx & 255;           // Wot[n][k] = Wo[k][n]
    Wot[idx] = (bf16_t)Wo[k * 128 + n];
  }
}

// ---------------------------------------------------------------------------
// Kernel 2: fused QKVG projection.  M=65536 rows, K=128, N=1024.
// One block = 16 rows; 8 waves each own 128 output cols (8 N-tiles).
// Epilogue scatters into attention-friendly layouts:
//   qb,kb,gb : [s][h][l][c]   (row-major, c contiguous)
//   vt       : [s][h][c][l]   (transposed so PV B-frags are contiguous)
// ---------------------------------------------------------------------------
__global__ void __launch_bounds__(256) proj_kernel(
    const bf16_t* __restrict__ xb, const bf16_t* __restrict__ Wt,
    const float* __restrict__ bg,
    bf16_t* __restrict__ qb, bf16_t* __restrict__ kb,
    bf16_t* __restrict__ vt, bf16_t* __restrict__ gb) {
  const int m0   = blockIdx.x * 16;
  const int wave = threadIdx.x >> 5;
  const int lane = threadIdx.x & 31;
  const int hl   = lane >> 4, l16 = lane & 15;

  __builtin_prefetch(xb + (size_t)(m0 + 16) * CIN, 0, 0);

  v16bf a[4];
#pragma unroll
  for (int kk = 0; kk < 4; ++kk) a[kk] = load_a_frag(xb, CIN, m0, kk * 32);

  const int nbase = wave * 128;
#pragma unroll
  for (int nt = 0; nt < 8; ++nt) {
    const int n0 = nbase + nt * 16;
    v8f acc = {};
#pragma unroll
    for (int kk = 0; kk < 4; ++kk)
      acc = wmma_bf16(a[kk], load_b_frag_kmajor(Wt, CIN, kk * 32, n0), acc);

    const int n     = n0 + l16;
    const int which = n >> 8;            // 0:q 1:k 2:v 3:g (uniform per wave)
    const int hc    = n & 255;
    const int h     = hc >> 5, c = hc & 31;
#pragma unroll
    for (int r = 0; r < 8; ++r) {
      const int m = m0 + r + hl * 8;
      const int s = m >> 8, l = m & 255;
      const size_t sh = (size_t)(s * NHEAD + h);
      const float p = acc[r];
      if (which == 0) {
        qb[(sh * 256 + l) * CH + c] = (bf16_t)(p * QK_SCALE);
      } else if (which == 1) {
        kb[(sh * 256 + l) * CH + c] = (bf16_t)p;
      } else if (which == 2) {
        vt[(sh * CH + c) * 256 + l] = (bf16_t)p;
      } else {
        const float gv = p + bg[hc];
        gb[(sh * 256 + l) * CH + c] = (bf16_t)(1.0f / (1.0f + __expf(-gv)));
      }
    }
  }
}

// ---------------------------------------------------------------------------
// Kernel 3: flash attention per (s,h). 2048 blocks, 256 threads = 8 waves.
// K-tile (256x32) and Vt-tile (32x256) staged in LDS by the Tensor Data
// Mover (one 1-D D# each, issued by wave 0, s_wait_tensorcnt 0); online
// softmax keeps the 256x256 logits entirely in registers per 16-query tile.
// ---------------------------------------------------------------------------
__global__ void __launch_bounds__(256) attn_kernel(
    const bf16_t* __restrict__ qb, const bf16_t* __restrict__ kb,
    const bf16_t* __restrict__ vtw, const float* __restrict__ bias,
    const unsigned char* __restrict__ amask, const bf16_t* __restrict__ gb,
    bf16_t* __restrict__ ob) {
  __shared__ bf16_t kt[256 * CH];          // [key][c]   16 KB
  __shared__ bf16_t vts[CH * 256];         // [c][key]   16 KB
  __shared__ float  maskadd[256];          // additive mask (0 or -3e38)
  __shared__ bf16_t pscr[8 * 16 * CH];     // per-wave prob scratch, 8 KB

  const int sh = blockIdx.x;               // s*8 + h
  const int s  = sh >> 3, h = sh & 7;
  const bf16_t* ksh = kb  + (size_t)sh * (256 * CH);
  const bf16_t* vsh = vtw + (size_t)sh * (CH * 256);

  // TDM staging: two contiguous 16 KB tiles, DMA'd by wave 0.
  if (threadIdx.x < 32) {
    tdm_load_1d_bf16(ksh, lds_addr_of(kt),  256 * CH);
    tdm_load_1d_bf16(vsh, lds_addr_of(vts), CH * 256);
    __builtin_amdgcn_s_wait_tensorcnt(0);
  }
  maskadd[threadIdx.x] = amask[s * 256 + threadIdx.x] ? 0.0f : -3.0e38f;
  __syncthreads();

  const int wave = threadIdx.x >> 5;
  const int lane = threadIdx.x & 31;
  const int hl = lane >> 4, l16 = lane & 15;
  const float* biash = bias + (size_t)h * (256 * 256);
  const bf16_t* qsh = qb + (size_t)sh * (256 * CH);
  const bf16_t* gsh = gb + (size_t)sh * (256 * CH);
  bf16_t* pw = pscr + wave * (16 * CH);

#pragma unroll
  for (int mt = 0; mt < 2; ++mt) {
    const int q0 = wave * 32 + mt * 16;
    const v16bf qf = load_a_frag(qsh, CH, q0, 0);   // 16x32: whole head dim
    v8f oacc0 = {}, oacc1 = {};
    float mrow[8], lrow[8];
#pragma unroll
    for (int r = 0; r < 8; ++r) { mrow[r] = -3.0e38f; lrow[r] = 0.0f; }

    for (int kc = 0; kc < 8; ++kc) {                // 32 keys per chunk
      const int k0 = kc * 32;
      // logits: q(16x32) * k^T(32x16) -> one WMMA per 16-key tile
      v8f t0 = wmma_bf16(qf, load_b_frag_kmajor(kt, CH, 0, k0), v8f{});
      v8f t1 = wmma_bf16(qf, load_b_frag_kmajor(kt, CH, 0, k0 + 16), v8f{});

#pragma unroll
      for (int r = 0; r < 8; ++r) {
        const int qm = q0 + r + hl * 8;
        float e0 = t0[r] + biash[(size_t)qm * 256 + k0 + l16]      + maskadd[k0 + l16];
        float e1 = t1[r] + biash[(size_t)qm * 256 + k0 + 16 + l16] + maskadd[k0 + 16 + l16];
        // row reduction across the 16 lanes holding this row (xor<16 stays in half)
        float cm = fmaxf(e0, e1);
#pragma unroll
        for (int off = 8; off >= 1; off >>= 1) cm = fmaxf(cm, __shfl_xor(cm, off, 32));
        const float mnew  = fmaxf(fmaxf(mrow[r], cm), -1.0e30f);
        const float scale = __expf(mrow[r] - mnew);
        const float p0 = __expf(e0 - mnew), p1 = __expf(e1 - mnew);
        float cs = p0 + p1;
#pragma unroll
        for (int off = 8; off >= 1; off >>= 1) cs += __shfl_xor(cs, off, 32);
        lrow[r] = lrow[r] * scale + cs;
        mrow[r] = mnew;
        oacc0[r] *= scale; oacc1[r] *= scale;
        // C-layout -> A-fragment reshuffle through per-wave LDS scratch
        const int m = r + hl * 8;
        pw[m * CH + l16]      = (bf16_t)p0;
        pw[m * CH + 16 + l16] = (bf16_t)p1;
      }
      __builtin_amdgcn_wave_barrier();   // LDS is in-order per wave; fence scheduler
      const v16bf pf = load_a_frag(pw, CH, 0, 0);   // probs as 16x32 A-matrix
      oacc0 = wmma_bf16(pf, load_b_frag_kmajor(vts, 256, k0, 0),  oacc0);
      oacc1 = wmma_bf16(pf, load_b_frag_kmajor(vts, 256, k0, 16), oacc1);
      __builtin_amdgcn_wave_barrier();
    }

    // normalize, gate, store to [s][l][h*32+c] for the output GEMM
#pragma unroll
    for (int r = 0; r < 8; ++r) {
      const float inv = 1.0f / lrow[r];
      const int qm = q0 + r + hl * 8;
      const float g0 = (float)gsh[qm * CH + l16];
      const float g1 = (float)gsh[qm * CH + 16 + l16];
      const size_t orow = ((size_t)(s * 256 + qm)) * (NHEAD * CH) + h * CH;
      ob[orow + l16]      = (bf16_t)(oacc0[r] * inv * g0);
      ob[orow + 16 + l16] = (bf16_t)(oacc1[r] * inv * g1);
    }
  }
}

// ---------------------------------------------------------------------------
// Kernel 4: output projection. out = ob(65536x256) @ Wo(256x128) + bo, fp32.
// One block = 16 rows; wave w owns N-tile w (8 waves x 16 = 128 cols).
// ---------------------------------------------------------------------------
__global__ void __launch_bounds__(256) outproj_kernel(
    const bf16_t* __restrict__ ob, const bf16_t* __restrict__ Wot,
    const float* __restrict__ bo, float* __restrict__ out) {
  const int m0   = blockIdx.x * 16;
  const int wave = threadIdx.x >> 5;
  const int lane = threadIdx.x & 31;
  const int hl = lane >> 4, l16 = lane & 15;
  const int n0 = wave * 16;

  __builtin_prefetch(ob + (size_t)(m0 + 16) * 256, 0, 0);

  v8f acc = {};
#pragma unroll
  for (int kk = 0; kk < 8; ++kk) {
    v16bf a = load_a_frag(ob, 256, m0, kk * 32);
    v16bf b = load_b_frag_kmajor(Wot, 256, kk * 32, n0);
    acc = wmma_bf16(a, b, acc);
  }
  const float bv = bo[n0 + l16];
#pragma unroll
  for (int r = 0; r < 8; ++r) {
    const int m = m0 + r + hl * 8;
    out[(size_t)m * CIN + n0 + l16] = acc[r] + bv;
  }
}

// ---------------------------------------------------------------------------
extern "C" void kernel_launch(void* const* d_in, const int* in_sizes, int n_in,
                              void* d_out, int out_size, void* d_ws, size_t ws_size,
                              hipStream_t stream) {
  const float* x    = (const float*)d_in[0];
  const float* bias = (const float*)d_in[1];
  const unsigned char* amask = (const unsigned char*)d_in[2];
  const float* Wq = (const float*)d_in[3];
  const float* Wk = (const float*)d_in[4];
  const float* Wv = (const float*)d_in[5];
  const float* Wo = (const float*)d_in[6];
  const float* bo = (const float*)d_in[7];
  const float* Wg = (const float*)d_in[8];
  const float* bg = (const float*)d_in[9];
  float* out = (float*)d_out;

  // workspace carve-out (~185 MB)
  char* ws = (char*)d_ws;
  size_t off = 0;
  auto carve = [&](size_t bytes) -> void* {
    void* p = ws + off;
    off += (bytes + 255) & ~(size_t)255;
    return p;
  };
  const size_t M = 65536;                         // S*S rows
  bf16_t* xb  = (bf16_t*)carve(M * CIN * 2);      // x in bf16
  bf16_t* Wt  = (bf16_t*)carve(1024 * 128 * 2);   // [Wq|Wk|Wv|Wg]^T
  bf16_t* Wot = (bf16_t*)carve(128 * 256 * 2);    // Wo^T
  bf16_t* qbuf = (bf16_t*)carve((size_t)NSH * 256 * CH * 2);
  bf16_t* kbuf = (bf16_t*)carve((size_t)NSH * 256 * CH * 2);
  bf16_t* vtbuf = (bf16_t*)carve((size_t)NSH * CH * 256 * 2);
  bf16_t* gbuf = (bf16_t*)carve((size_t)NSH * 256 * CH * 2);
  bf16_t* obuf = (bf16_t*)carve(M * (NHEAD * CH) * 2);
  (void)ws_size; (void)in_sizes; (void)n_in; (void)out_size;

  cast_x_kernel<<<(int)(M * CIN / 4 / 256), 256, 0, stream>>>(x, xb, (int)(M * CIN / 4));
  prep_w_kernel<<<512, 256, 0, stream>>>(Wq, Wk, Wv, Wg, Wo, Wt, Wot);
  proj_kernel<<<(int)(M / 16), 256, 0, stream>>>(xb, Wt, bg, qbuf, kbuf, vtbuf, gbuf);
  attn_kernel<<<NSH, 256, 0, stream>>>(qbuf, kbuf, vtbuf, bias, amask, gbuf, obuf);
  outproj_kernel<<<(int)(M / 16), 256, 0, stream>>>(obuf, Wot, bo, out);
}
